// ContrastiveWSD_28802050687659
// MI455X (gfx1250) — matrix operand order repaired and checked by reading, not compiled, not run on hardware
//
#include <hip/hip_runtime.h>

#define DIM 768
#define NEG_SLOPE_C 0.2f
#define LN_EPS 1e-5f

typedef __attribute__((ext_vector_type(16))) __bf16 bf16x16;
typedef __attribute__((ext_vector_type(8)))  __bf16 bf16x8;
typedef __attribute__((ext_vector_type(8)))  float  f32x8;

// ---------- helpers ----------
__device__ __forceinline__ unsigned fkey(float f) {
    unsigned b = __float_as_uint(f);
    return (b & 0x80000000u) ? ~b : (b | 0x80000000u);   // order-preserving float->uint
}
__device__ __forceinline__ float funkey(unsigned u) {
    unsigned b = (u & 0x80000000u) ? (u ^ 0x80000000u) : ~u;
    return __uint_as_float(b);
}
__device__ __forceinline__ float lrelu(float x) { return x > 0.f ? x : NEG_SLOPE_C * x; }
__device__ __forceinline__ float gelu_exact(float x) {
    return 0.5f * x * (1.f + erff(x * 0.70710678118654752f));
}

// CDNA5 async global->LDS copy (ASYNCcnt-tracked) + its wait.
// byte_off must fold to a compile-time constant (signed 24-bit instruction offset):
// keeps the base address register loop-invariant -> no 64-bit VALU adds (and no
// WMMA->VALU hazard v_nops) inside the unrolled GEMM K-loop.
__device__ __forceinline__ void async_load_b128(unsigned lds_addr, const void* gaddr,
                                                int byte_off) {
    asm volatile("global_load_async_to_lds_b128 %0, %1, off offset:%2"
                 :: "v"(lds_addr), "v"(gaddr), "i"(byte_off) : "memory");
}
__device__ __forceinline__ void wait_asynccnt0() {
    asm volatile("s_wait_asynccnt 0x0" ::: "memory");
}

// ---------- weight transpose + bf16 convert:  WT[n*768+k] = bf16(W[k*768+n]) ----------
__global__ void wconvert_kernel(const float* __restrict__ W, __bf16* __restrict__ WT) {
    int idx = blockIdx.x * 256 + threadIdx.x;   // over DIM*DIM
    int n = idx / DIM, k = idx % DIM;
    WT[idx] = (__bf16)W[k * DIM + n];
}

// ---------- layernorm + exact gelu, one block (256 thr) per row ----------
template <typename OUT>
__global__ __launch_bounds__(256) void ln_gelu_kernel(const float* __restrict__ X,
                                                      const float* __restrict__ g,
                                                      const float* __restrict__ b,
                                                      OUT* __restrict__ Y) {
    int row = blockIdx.x, t = threadIdx.x;
    const float* x = X + (size_t)row * DIM;
    float x0 = x[t], x1 = x[t + 256], x2 = x[t + 512];
    __shared__ float red[8];
    float s = x0 + x1 + x2;
#pragma unroll
    for (int o = 16; o; o >>= 1) s += __shfl_xor(s, o, 32);
    if ((t & 31) == 0) red[t >> 5] = s;
    __syncthreads();
    float mean = (red[0] + red[1] + red[2] + red[3] + red[4] + red[5] + red[6] + red[7]) * (1.f / DIM);
    float d0 = x0 - mean, d1 = x1 - mean, d2 = x2 - mean;
    float q = d0 * d0 + d1 * d1 + d2 * d2;
    __syncthreads();
#pragma unroll
    for (int o = 16; o; o >>= 1) q += __shfl_xor(q, o, 32);
    if ((t & 31) == 0) red[t >> 5] = q;
    __syncthreads();
    float var = (red[0] + red[1] + red[2] + red[3] + red[4] + red[5] + red[6] + red[7]) * (1.f / DIM);
    float rs = rsqrtf(var + LN_EPS);
    OUT* y = Y + (size_t)row * DIM;
    y[t]       = (OUT)gelu_exact(d0 * rs * g[t]       + b[t]);
    y[t + 256] = (OUT)gelu_exact(d1 * rs * g[t + 256] + b[t + 256]);
    y[t + 512] = (OUT)gelu_exact(d2 * rs * g[t + 512] + b[t + 512]);
}

// ---------- bf16 WMMA GEMM: C[M,768] = A[M,768] @ W + bias ; WT is W^T (bf16) ----------
// Block = 256 thr (8 waves). Block tile: 256 rows x 64 cols. Wave tile: 32 rows x 64 cols.
// B tile (32K x 64 cols, 4KB) staged in LDS via async global->LDS, double buffered.
// A fragments register double-buffered. 8 WMMAs per wave per K-step, 24 K-steps.
// K-loop fully unrolled: prefetch-buffer rotation is pure renaming; async source
// offsets are instruction immediates.
#define WMMA_BF16(ac, av, bv) \
    ac = __builtin_amdgcn_wmma_f32_16x16x32_bf16(false, av, false, bv, (short)0, ac, false, false)

__global__ __launch_bounds__(256) void gemm_bf16_wmma_kernel(const __bf16* __restrict__ A,
                                                             const __bf16* __restrict__ WT,
                                                             const float* __restrict__ bias,
                                                             float* __restrict__ C, int M) {
    __shared__ __align__(32) unsigned char ldsB[2][4096];   // [buf][col(64)][k(32)] bf16

    const int wave = threadIdx.x >> 5;
    const int lane = threadIdx.x & 31;
    const int row0 = blockIdx.x * 256 + wave * 32;
    const int col0 = blockIdx.y * 64;
    const int hi   = lane >> 4;          // K-half select: lane half hi holds k = 16*hi + s (A and B)
    const int l16  = lane & 15;
    // Never early-exit (barriers!): clamp compute rows, guard stores.
    const bool valid = (row0 + 32 <= M);
    const int rowA  = valid ? row0 : (M - 32);

    // --- async B-tile feeder: thread t loads 16B: col = t>>2, k-quarter = t&3 ---
    const int tcol = threadIdx.x >> 2;   // 0..63
    const int tkq  = threadIdx.x & 3;    // 0..3
    const __bf16* gsrc = WT + (size_t)(col0 + tcol) * DIM + tkq * 8;
    const unsigned ldsBase = (unsigned)(size_t)(&ldsB[0][0]);   // low 32 bits = LDS offset
    const unsigned ldsDst  = ldsBase + (unsigned)(tcol * 64 + tkq * 16);

    async_load_b128(ldsDst, gsrc, 0);                   // K-step 0 -> buf 0

    // --- A fragment pointers (two 16-row tiles) ---
    const __bf16* arow0 = A + (size_t)(rowA + l16) * DIM + hi * 16;
    const __bf16* arow1 = arow0 + (size_t)16 * DIM;
    bf16x8 a0lo = *(const bf16x8*)(arow0);
    bf16x8 a0hi = *(const bf16x8*)(arow0 + 8);
    bf16x8 a1lo = *(const bf16x8*)(arow1);
    bf16x8 a1hi = *(const bf16x8*)(arow1 + 8);

    f32x8 acc0 = {}, acc1 = {}, acc2 = {}, acc3 = {};
    f32x8 acc4 = {}, acc5 = {}, acc6 = {}, acc7 = {};

    wait_asynccnt0();
    __syncthreads();

#pragma unroll
    for (int i = 0; i < 24; ++i) {
        const int kk = i * 32;
        const int cur = i & 1;
        bf16x8 na0lo, na0hi, na1lo, na1hi;
        if (i < 23) {
            // kick off next B tile into the other LDS buffer (overwrites buffer whose
            // readers finished before last barrier) and prefetch next A fragments.
            async_load_b128(ldsDst + (unsigned)(((i + 1) & 1) * 4096), gsrc,
                            (kk + 32) * (int)sizeof(__bf16));
            na0lo = *(const bf16x8*)(arow0 + kk + 32);
            na0hi = *(const bf16x8*)(arow0 + kk + 40);
            na1lo = *(const bf16x8*)(arow1 + kk + 32);
            na1hi = *(const bf16x8*)(arow1 + kk + 40);
        }
        bf16x16 a0 = __builtin_shufflevector(a0lo, a0hi, 0, 1, 2, 3, 4, 5, 6, 7,
                                             8, 9, 10, 11, 12, 13, 14, 15);
        bf16x16 a1 = __builtin_shufflevector(a1lo, a1hi, 0, 1, 2, 3, 4, 5, 6, 7,
                                             8, 9, 10, 11, 12, 13, 14, 15);
        // B fragments from LDS: col = 16*t + l16, offset = col*64 + hi*32 bytes
        const unsigned char* bb = &ldsB[cur][l16 * 64 + hi * 32];
        bf16x16 b0 = *(const bf16x16*)(bb);
        bf16x16 b1 = *(const bf16x16*)(bb + 16 * 64);
        bf16x16 b2 = *(const bf16x16*)(bb + 32 * 64);
        bf16x16 b3 = *(const bf16x16*)(bb + 48 * 64);

        WMMA_BF16(acc0, a0, b0); WMMA_BF16(acc1, a0, b1);
        WMMA_BF16(acc2, a0, b2); WMMA_BF16(acc3, a0, b3);
        WMMA_BF16(acc4, a1, b0); WMMA_BF16(acc5, a1, b1);
        WMMA_BF16(acc6, a1, b2); WMMA_BF16(acc7, a1, b3);

        if (i < 23) {
            a0lo = na0lo; a0hi = na0hi; a1lo = na1lo; a1hi = na1hi;
            wait_asynccnt0();       // next B buffer resident (overlapped with the 8 WMMAs)
            __syncthreads();        // all waves done reading old buffer / feeder data ready
        }
    }

    if (valid) {
        const float bb0 = bias[col0 + l16];
        const float bb1 = bias[col0 + l16 + 16];
        const float bb2 = bias[col0 + l16 + 32];
        const float bb3 = bias[col0 + l16 + 48];
#pragma unroll
        for (int v = 0; v < 8; ++v) {
            int r = row0 + v + hi * 8;           // C/D layout: vgpr v -> row (+8 for hi lanes)
            float* c0 = C + (size_t)r * DIM + col0 + l16;
            c0[0]  = acc0[v] + bb0;
            c0[16] = acc1[v] + bb1;
            c0[32] = acc2[v] + bb2;
            c0[48] = acc3[v] + bb3;
            float* c1 = c0 + (size_t)16 * DIM;   // second 16-row tile
            c1[0]  = acc4[v] + bb0;
            c1[16] = acc5[v] + bb1;
            c1[32] = acc6[v] + bb2;
            c1[48] = acc7[v] + bb3;
        }
    }
}

// ---------- per-layer node init: H = bias (broadcast), emax key=0, denom=0 ----------
__global__ void init_nodes_kernel(float* __restrict__ H, const float* __restrict__ bias,
                                  unsigned* __restrict__ emax_u, float* __restrict__ denom,
                                  int N) {
    size_t idx = (size_t)blockIdx.x * 256 + threadIdx.x;
    if (idx >= (size_t)N * DIM) return;
    int d = (int)(idx % DIM);
    H[idx] = bias[d];
    if (d == 0) {
        int n = (int)(idx / DIM);
        emax_u[n] = 0u;
        denom[n]  = 0.f;
    }
}

// ---------- pass 1: e = sum(leaky_relu(xl[src]+xr[dst])*att); segment max ----------
__global__ __launch_bounds__(256) void edge_e_kernel(const float* __restrict__ xl,
                                                     const float* __restrict__ xr,
                                                     const float* __restrict__ att,
                                                     const int* __restrict__ edges,
                                                     float* __restrict__ e_out,
                                                     unsigned* __restrict__ emax_u,
                                                     int E, int N) {
    int eid = blockIdx.x * 8 + (threadIdx.x >> 5);   // wave per edge
    if (eid >= E + N) return;
    int lane = threadIdx.x & 31;
    int src, dst;
    if (eid < E) { src = edges[eid]; dst = edges[E + eid]; }
    else         { src = eid - E;    dst = src; }      // self loops
    const float4* pl = (const float4*)(xl + (size_t)src * DIM);
    const float4* pr = (const float4*)(xr + (size_t)dst * DIM);
    const float4* pa = (const float4*)att;
    float s = 0.f;
#pragma unroll
    for (int j = 0; j < 6; ++j) {
        int i = lane + j * 32;
        float4 a = pl[i], b = pr[i], c = pa[i];
        s += lrelu(a.x + b.x) * c.x + lrelu(a.y + b.y) * c.y +
             lrelu(a.z + b.z) * c.z + lrelu(a.w + b.w) * c.w;
    }
#pragma unroll
    for (int o = 16; o; o >>= 1) s += __shfl_xor(s, o, 32);
    if (lane == 0) {
        e_out[eid] = s;
        atomicMax(&emax_u[dst], fkey(s));
    }
}

// ---------- pass 2: ex = exp(e - emax[dst]); denom[dst] += ex ----------
__global__ void edge_ex_kernel(const float* __restrict__ e_in, const int* __restrict__ edges,
                               const unsigned* __restrict__ emax_u, float* __restrict__ denom,
                               float* __restrict__ ex_out, int E, int N) {
    int eid = blockIdx.x * 256 + threadIdx.x;
    if (eid >= E + N) return;
    int dst = (eid < E) ? edges[E + eid] : (eid - E);
    float em = funkey(emax_u[dst]);
    if (!__builtin_isfinite(em)) em = 0.f;      // mirrors reference guard
    float ex = __expf(e_in[eid] - em);
    ex_out[eid] = ex;
    atomicAdd(&denom[dst], ex);
}

// ---------- pass 3: H[dst] += (ex/denom[dst]) * xl[src] ----------
__global__ __launch_bounds__(256) void edge_scatter_kernel(const float* __restrict__ xl,
                                                           const float* __restrict__ ex_in,
                                                           const float* __restrict__ denom,
                                                           const int* __restrict__ edges,
                                                           float* __restrict__ H,
                                                           int E, int N) {
    int eid = blockIdx.x * 8 + (threadIdx.x >> 5);
    if (eid >= E + N) return;
    int lane = threadIdx.x & 31;
    int src, dst;
    if (eid < E) { src = edges[eid]; dst = edges[E + eid]; }
    else         { src = eid - E;    dst = src; }
    float alpha = ex_in[eid] / denom[dst];
    const float4* pl = (const float4*)(xl + (size_t)src * DIM);
    float* out = H + (size_t)dst * DIM;
#pragma unroll
    for (int j = 0; j < 6; ++j) {
        int i = lane + j * 32;
        float4 v = pl[i];
        atomicAdd(out + i * 4 + 0, alpha * v.x);
        atomicAdd(out + i * 4 + 1, alpha * v.y);
        atomicAdd(out + i * 4 + 2, alpha * v.z);
        atomicAdd(out + i * 4 + 3, alpha * v.w);
    }
}

// ---------- host ----------
extern "C" void kernel_launch(void* const* d_in, const int* in_sizes, int n_in,
                              void* d_out, int out_size, void* d_ws, size_t ws_size,
                              hipStream_t stream) {
    (void)n_in; (void)ws_size;
    const float* emb   = (const float*)d_in[0];
    const float* gx    = (const float*)d_in[1];
    const float* ln1g  = (const float*)d_in[2];
    const float* ln1b  = (const float*)d_in[3];
    const float* Wl1   = (const float*)d_in[4];
    const float* bl1   = (const float*)d_in[5];
    const float* Wr1   = (const float*)d_in[6];
    const float* br1   = (const float*)d_in[7];
    const float* att1  = (const float*)d_in[8];
    const float* bias1 = (const float*)d_in[9];
    const float* ln2g  = (const float*)d_in[10];
    const float* ln2b  = (const float*)d_in[11];
    const float* Wl2   = (const float*)d_in[12];
    const float* bl2   = (const float*)d_in[13];
    const float* Wr2   = (const float*)d_in[14];
    const float* br2   = (const float*)d_in[15];
    const float* att2  = (const float*)d_in[16];
    const float* bias2 = (const float*)d_in[17];
    const float* ln3g  = (const float*)d_in[18];
    const float* ln3b  = (const float*)d_in[19];
    const int*   edges = (const int*)d_in[20];

    const int N        = in_sizes[1] / DIM;        // 20000
    const int E        = in_sizes[20] / 2;         // 160000
    const int embRows  = in_sizes[0] / DIM;        // 64
    const int rows_out = out_size / DIM - embRows; // 4096

    // ---- workspace layout ----
    char* ws = (char*)d_ws;
    size_t off = 0;
    auto carve = [&](size_t bytes) -> char* {
        char* p = ws + off;
        off = (off + bytes + 255) & ~(size_t)255;
        return p;
    };
    const size_t wbytes = (size_t)DIM * DIM * sizeof(__bf16);
    __bf16* WTl1 = (__bf16*)carve(wbytes);
    __bf16* WTr1 = (__bf16*)carve(wbytes);
    __bf16* WTl2 = (__bf16*)carve(wbytes);
    __bf16* WTr2 = (__bf16*)carve(wbytes);
    __bf16* Abf  = (__bf16*)carve((size_t)N * DIM * sizeof(__bf16));
    float*  xl   = (float*)carve((size_t)N * DIM * sizeof(float));
    float*  xr   = (float*)carve((size_t)N * DIM * sizeof(float));
    float*  Hb   = (float*)carve((size_t)N * DIM * sizeof(float));
    float*  e_ws = (float*)carve((size_t)(E + N) * sizeof(float));
    float*  ex_ws= (float*)carve((size_t)(E + N) * sizeof(float));
    unsigned* emax_u = (unsigned*)carve((size_t)N * sizeof(unsigned));
    float*    denom  = (float*)carve((size_t)N * sizeof(float));

    dim3 blk(256);

    // 1) weights -> transposed bf16 (one-time per launch; tiny vs edge traffic)
    int wblocks = (DIM * DIM) / 256;
    wconvert_kernel<<<wblocks, blk, 0, stream>>>(Wl1, WTl1);
    wconvert_kernel<<<wblocks, blk, 0, stream>>>(Wr1, WTr1);
    wconvert_kernel<<<wblocks, blk, 0, stream>>>(Wl2, WTl2);
    wconvert_kernel<<<wblocks, blk, 0, stream>>>(Wr2, WTr2);

    const dim3 ggrid((N + 255) / 256, DIM / 64);
    const int T = E + N;
    const unsigned nodeBlocks = (unsigned)(((size_t)N * DIM + 255) / 256);

    auto run_gat = [&](const __bf16* WTl, const float* bl, const __bf16* WTr, const float* br,
                       const float* att, const float* bias) {
        gemm_bf16_wmma_kernel<<<ggrid, blk, 0, stream>>>(Abf, WTl, bl, xl, N);
        gemm_bf16_wmma_kernel<<<ggrid, blk, 0, stream>>>(Abf, WTr, br, xr, N);
        init_nodes_kernel<<<nodeBlocks, blk, 0, stream>>>(Hb, bias, emax_u, denom, N);
        edge_e_kernel<<<(T + 7) / 8, blk, 0, stream>>>(xl, xr, att, edges, e_ws, emax_u, E, N);
        edge_ex_kernel<<<(T + 255) / 256, blk, 0, stream>>>(e_ws, edges, emax_u, denom, ex_ws, E, N);
        edge_scatter_kernel<<<(T + 7) / 8, blk, 0, stream>>>(xl, ex_ws, denom, edges, Hb, E, N);
    };

    // stage 1: LN+GELU on input -> bf16 activations
    ln_gelu_kernel<__bf16><<<N, blk, 0, stream>>>(gx, ln1g, ln1b, Abf);
    // GAT layer 1 -> Hb
    run_gat(WTl1, bl1, WTr1, br1, att1, bias1);
    // stage 2: LN+GELU -> bf16 activations
    ln_gelu_kernel<__bf16><<<N, blk, 0, stream>>>(Hb, ln2g, ln2b, Abf);
    // GAT layer 2 -> Hb (reuse; previous contents fully consumed)
    run_gat(WTl2, bl2, WTr2, br2, att2, bias2);

    // outputs: passthrough embeddings + final LN+GELU on first rows_out rows
    hipMemcpyAsync(d_out, emb, (size_t)embRows * DIM * sizeof(float),
                   hipMemcpyDeviceToDevice, stream);
    ln_gelu_kernel<float><<<rows_out, blk, 0, stream>>>(Hb, ln3g, ln3b,
                                                        (float*)d_out + (size_t)embRows * DIM);
}
